// DetLoss_21466246545884
// MI455X (gfx1250) — compile-verified, matrix operand order checked by reading, not run on hardware
//
#include <hip/hip_runtime.h>
#include <hip/hip_bf16.h>

// ---------------------------------------------------------------------------
// YOLOv5-style detection loss on gfx1250.
// B=16, NA=3, NC=80, NO=85, grids 80/40/20. Output: single f32 scalar.
// Pipeline (all on `stream`, deterministic, no float atomics):
//   K0 zero_ws      : clear tobj keys + int accumulators
//   K1 entry_kernel : per matched target -> CIoU (lbox partial), cls BCE
//                     partial, packed atomicMax scatter of clip(iou,0)
//   K2 obj_kernel   : dense BCE(p[...,4], tobj) with per-wave WMMA reduction
//                     (V_WMMA_F32_16X16X4_F32 against an all-ones B matrix)
//   K3 finalize     : fixed-order reduction of all partials -> loss scalar
// ---------------------------------------------------------------------------

#define NAc 3
#define NCc 80
#define NOc 85
#define OBJ_BLOCKS 512
#define OBJ_THREADS 256

typedef float v2f  __attribute__((ext_vector_type(2)));
typedef float v8f  __attribute__((ext_vector_type(8)));
typedef _Float16 v16h __attribute__((ext_vector_type(16)));

__device__ __forceinline__ float softplusf(float x) {
    // log(1+e^x), stable
    return fmaxf(x, 0.0f) + log1pf(expf(-fabsf(x)));
}
__device__ __forceinline__ float sigmoidf_(float x) {
    return 1.0f / (1.0f + expf(-x));
}

// ---- wave32 all-lane sum using the CDNA5 matrix pipe ----------------------
// A: per-lane scalar in K-slot 0 (16x4 f32 A layout: lane L<16 -> (M=L,K=0),
// lane L>=16 -> (M=L-16,K=2)); B = ones(4x16) => D[m][n] = s_m + s_{m+16}.
// Each lane then sums its 8 accumulator rows (its M-half of one column);
// lane0 + lane16 give the full 32-lane sum. EXEC must be all ones here.
__device__ __forceinline__ float wave_sum_wmma(float s) {
#if __has_builtin(__builtin_amdgcn_wmma_f32_16x16x4_f32)
    v2f a; a[0] = s; a[1] = 0.0f;
    v2f b; b[0] = 1.0f; b[1] = 1.0f;
    v8f c = {};
    c = __builtin_amdgcn_wmma_f32_16x16x4_f32(false, a, false, b,
                                              (short)0, c, false, false);
#else
    // fallback: probe-confirmed f16 WMMA (s is O(1) here; f16 rounding of the
    // per-lane partial is far below validation tolerance)
    v16h a = {}; a[0] = (_Float16)s;
    v16h b;
#pragma unroll
    for (int i = 0; i < 16; ++i) b[i] = (_Float16)1.0f;
    v8f c = {};
    c = __builtin_amdgcn_wmma_f32_16x16x32_f16(false, a, false, b,
                                               (short)0, c, false, false);
#endif
    float half_sum = c[0] + c[1] + c[2] + c[3] + c[4] + c[5] + c[6] + c[7];
    // lanes 0..15 hold sum over M=0..7 of a column; lanes 16..31 hold M=8..15
    return __shfl(half_sum, 0, 32) + __shfl(half_sum, 16, 32);
}

// ---------------------------------------------------------------------------
__global__ void zero_ws_kernel(unsigned long long* __restrict__ keys, int nkeys,
                               int* __restrict__ iacc) {
    int tid = blockIdx.x * blockDim.x + threadIdx.x;
    int stride = gridDim.x * blockDim.x;
    for (int i = tid; i < nkeys; i += stride) keys[i] = 0ull;
    if (blockIdx.x == 0 && threadIdx.x < 16) iacc[threadIdx.x] = 0;
}

// ---------------------------------------------------------------------------
__global__ void entry_kernel(const float* __restrict__ p,
                             const float* __restrict__ tbox,
                             const float* __restrict__ anch,
                             const int* __restrict__ bi,
                             const int* __restrict__ ai,
                             const int* __restrict__ gji,
                             const int* __restrict__ gii,
                             const int* __restrict__ tci,
                             unsigned long long* __restrict__ keys,
                             float* __restrict__ lbox_part,
                             float* __restrict__ lcls_part,
                             int* __restrict__ keep_acc,
                             int n, int gh, int gw, int lvl) {
    int e = blockIdx.x * blockDim.x + threadIdx.x;
    if (e >= n) return;

    int b = bi[e], a = ai[e], gj = gji[e], gix = gii[e], tc = tci[e];
    long long cell = ((((long long)b * NAc + a) * gh + gj) * gw + gix);
    const float* pos = p + cell * NOc;

    // decode box
    float px = sigmoidf_(pos[0]) * 2.0f - 0.5f;
    float py = sigmoidf_(pos[1]) * 2.0f - 0.5f;
    float sw = sigmoidf_(pos[2]) * 2.0f;
    float sh = sigmoidf_(pos[3]) * 2.0f;
    float pw = sw * sw * anch[2 * e + 0];
    float ph = sh * sh * anch[2 * e + 1];
    float tx = tbox[4 * e + 0], ty = tbox[4 * e + 1];
    float tw = tbox[4 * e + 2], th = tbox[4 * e + 3];

    // CIoU
    const float eps = 1e-7f;
    float b1x1 = px - pw * 0.5f, b1x2 = px + pw * 0.5f;
    float b1y1 = py - ph * 0.5f, b1y2 = py + ph * 0.5f;
    float b2x1 = tx - tw * 0.5f, b2x2 = tx + tw * 0.5f;
    float b2y1 = ty - th * 0.5f, b2y2 = ty + th * 0.5f;
    float iw = fmaxf(fminf(b1x2, b2x2) - fmaxf(b1x1, b2x1), 0.0f);
    float ih = fmaxf(fminf(b1y2, b2y2) - fmaxf(b1y1, b2y1), 0.0f);
    float inter = iw * ih;
    float uni = pw * ph + tw * th - inter + eps;
    float iou = inter / uni;
    float cw = fmaxf(b1x2, b2x2) - fminf(b1x1, b2x1);
    float ch = fmaxf(b1y2, b2y2) - fminf(b1y1, b2y1);
    float c2 = cw * cw + ch * ch + eps;
    float dx = b2x1 + b2x2 - b1x1 - b1x2;
    float dy = b2y1 + b2y2 - b1y1 - b1y2;
    float rho2 = (dx * dx + dy * dy) * 0.25f;
    float dat = atanf(tw / (th + eps)) - atanf(pw / (ph + eps));
    float v = 0.40528473456935108577f * dat * dat;  // 4/pi^2
    float alpha = v / (v - iou + 1.0f + eps);
    float ciou = iou - (rho2 / c2 + v * alpha);

    lbox_part[e] = 1.0f - ciou;

    // scatter clip(ciou,0) into dense tobj; deterministic duplicate
    // resolution: max iou wins (ties -> larger entry index)
    float iou_t = fmaxf(ciou, 0.0f);
    unsigned long long key =
        (((unsigned long long)__float_as_uint(iou_t)) << 32) |
        (unsigned long long)(unsigned)(e + 1);
    atomicMax(&keys[cell], key);

    // class BCE with one-hot target: sum_c softplus(x_c) - x_{tc-1}
    float csum = 0.0f;
    const float* pcls = pos + 5;
#pragma unroll 4
    for (int c = 0; c < NCc; ++c) csum += softplusf(pcls[c]);
    int cidx = (tc - 1) > 0 ? (tc - 1) : 0;
    csum -= pcls[cidx];
    bool keep = (tc > 0);
    lcls_part[e] = keep ? csum : 0.0f;
    if (keep) atomicAdd(&keep_acc[lvl], 1);  // integer atomics: deterministic
}

// ---------------------------------------------------------------------------
// Dense objectness BCE; tobj unpacked from the high 32 bits of the scatter
// keys (0 => no target => tobj = 0). Per-wave reduction via WMMA, cross-wave
// via LDS, one deterministic slot per block.
__global__ void obj_kernel(const float* __restrict__ p,
                           const unsigned long long* __restrict__ keys,
                           int ncells, float* __restrict__ obj_part) {
    int tid = blockIdx.x * blockDim.x + threadIdx.x;
    int stride = gridDim.x * blockDim.x;
    float s = 0.0f;
    for (int c = tid; c < ncells; c += stride) {
        float x = p[(size_t)c * NOc + 4];
        unsigned long long k = keys[c];
        float t = __uint_as_float((unsigned)(k >> 32));
        s += softplusf(x) - t * x;  // BCE(x, t)
    }
    // uniform control flow from here on: EXEC all ones for WMMA
    float wtot = wave_sum_wmma(s);

    __shared__ float lds[OBJ_THREADS / 32];
    int lane = threadIdx.x & 31;
    int wid = threadIdx.x >> 5;
    if (lane == 0) lds[wid] = wtot;
    __syncthreads();
    if (threadIdx.x == 0) {
        float bsum = 0.0f;
#pragma unroll
        for (int i = 0; i < OBJ_THREADS / 32; ++i) bsum += lds[i];
        obj_part[blockIdx.x] = bsum;
    }
}

// ---------------------------------------------------------------------------
__device__ float block_sum_det(const float* __restrict__ arr, int n,
                               float* __restrict__ lds) {
    float s = 0.0f;
    for (int i = threadIdx.x; i < n; i += blockDim.x) s += arr[i];
    lds[threadIdx.x] = s;
    __syncthreads();
    float r = 0.0f;
    if (threadIdx.x == 0)
        for (int i = 0; i < (int)blockDim.x; ++i) r += lds[i];
    __syncthreads();
    return r;  // valid in thread 0
}

__global__ void finalize_kernel(const float* __restrict__ lb0,
                                const float* __restrict__ lb1,
                                const float* __restrict__ lb2,
                                const float* __restrict__ lc0,
                                const float* __restrict__ lc1,
                                const float* __restrict__ lc2,
                                const float* __restrict__ objp,
                                const int* __restrict__ iacc,
                                float* __restrict__ out,
                                int n0, int n1, int n2) {
    __shared__ float lds[256];
    const float* lbs[3] = {lb0, lb1, lb2};
    const float* lcs[3] = {lc0, lc1, lc2};
    int ns[3] = {n0, n1, n2};
    const float bal[3] = {4.0f, 1.0f, 0.4f};
    const float cells[3] = {16.0f * 3 * 80 * 80, 16.0f * 3 * 40 * 40,
                            16.0f * 3 * 20 * 20};
    float lbox = 0.0f, lobj = 0.0f, lcls = 0.0f;
    for (int i = 0; i < 3; ++i) {
        float sb = block_sum_det(lbs[i], ns[i], lds);
        float sc = block_sum_det(lcs[i], ns[i], lds);
        float so = block_sum_det(objp + i * OBJ_BLOCKS, OBJ_BLOCKS, lds);
        if (threadIdx.x == 0) {
            lbox += sb / (float)ns[i];
            int k = iacc[i]; if (k < 1) k = 1;
            lcls += sc / ((float)k * (float)NCc);
            lobj += (so / cells[i]) * bal[i];
        }
    }
    if (threadIdx.x == 0)
        out[0] = (lbox * 0.05f + lobj * 1.0f + lcls * 0.05f) * 16.0f;
}

// ---------------------------------------------------------------------------
extern "C" void kernel_launch(void* const* d_in, const int* in_sizes, int n_in,
                              void* d_out, int out_size, void* d_ws,
                              size_t ws_size, hipStream_t stream) {
    // dict order per level: p, tbox, anch, b, a, gj, gi, tcls
    const float *p[3], *tb[3], *an[3];
    const int *bb[3], *aa[3], *gj[3], *gi[3], *tc[3];
    int n[3];
    for (int i = 0; i < 3; ++i) {
        int base = 8 * i;
        p[i]  = (const float*)d_in[base + 0];
        tb[i] = (const float*)d_in[base + 1];
        an[i] = (const float*)d_in[base + 2];
        bb[i] = (const int*)d_in[base + 3];
        aa[i] = (const int*)d_in[base + 4];
        gj[i] = (const int*)d_in[base + 5];
        gi[i] = (const int*)d_in[base + 6];
        tc[i] = (const int*)d_in[base + 7];
        n[i]  = in_sizes[base + 3];
    }
    const int gh[3] = {80, 40, 20}, gw[3] = {80, 40, 20};
    const int cells[3] = {16 * 3 * 80 * 80, 16 * 3 * 40 * 40, 16 * 3 * 20 * 20};
    const int totc = cells[0] + cells[1] + cells[2];  // 403200

    // workspace carve-out
    char* ws = (char*)d_ws;
    int* iacc = (int*)ws;                                   // 16 ints
    size_t off = 64;
    unsigned long long* keys = (unsigned long long*)(ws + off);
    off += (size_t)totc * sizeof(unsigned long long);
    float* lboxp[3]; float* lclsp[3];
    for (int i = 0; i < 3; ++i) { lboxp[i] = (float*)(ws + off); off += (size_t)n[i] * 4; }
    for (int i = 0; i < 3; ++i) { lclsp[i] = (float*)(ws + off); off += (size_t)n[i] * 4; }
    float* objp = (float*)(ws + off);                       // 3 * OBJ_BLOCKS

    zero_ws_kernel<<<256, 256, 0, stream>>>(keys, totc, iacc);

    unsigned long long* kptr = keys;
    for (int i = 0; i < 3; ++i) {
        int blocks = (n[i] + 127) / 128; if (blocks < 1) blocks = 1;
        entry_kernel<<<blocks, 128, 0, stream>>>(
            p[i], tb[i], an[i], bb[i], aa[i], gj[i], gi[i], tc[i],
            kptr, lboxp[i], lclsp[i], iacc, n[i], gh[i], gw[i], i);
        kptr += cells[i];
    }

    kptr = keys;
    for (int i = 0; i < 3; ++i) {
        obj_kernel<<<OBJ_BLOCKS, OBJ_THREADS, 0, stream>>>(
            p[i], kptr, cells[i], objp + i * OBJ_BLOCKS);
        kptr += cells[i];
    }

    finalize_kernel<<<1, 256, 0, stream>>>(
        lboxp[0], lboxp[1], lboxp[2], lclsp[0], lclsp[1], lclsp[2],
        objp, iacc, (float*)d_out, n[0], n[1], n[2]);
}
